// BrainLayer_40982577938844
// MI455X (gfx1250) — matrix operand order repaired and checked by reading, not compiled, not run on hardware
//
#include <hip/hip_runtime.h>
#include <hip/hip_bf16.h>

// ---------------------------------------------------------------------------
// Echo-state reservoir on MI455X (gfx1250).
//   r_{t} = (1-g) r_{t-1} + g*tanh( r_{t-1} W_rec^T + x_t (in_cor W_in)^T + b )
//   out_t = r_t out_cor^T
// Recurrent + output GEMMs run on v_wmma_f32_16x16x32_bf16 (f32 accumulate).
// Weights are packed once per launch into "fragment-major" bf16 layout so the
// B operand of every WMMA is two contiguous global_load_b128 per lane.
// The 512 sequential steps use 4-way split-K per tile (LDS reduction) to cut
// the per-step WMMA dependency chain from 68 to 17 and quadruple wave count.
// ---------------------------------------------------------------------------

#define RES_N   2048
#define FEAT_F  128
#define BATCH_B 32
#define TIME_T  512
#define GAMMA_F 0.95f

typedef __attribute__((ext_vector_type(16))) __bf16 v16bf;
typedef __attribute__((ext_vector_type(8)))  float  v8f;

union Frag {
    v16bf        bf;
    uint4        q[2];
    unsigned int u[8];
};

__device__ __forceinline__ unsigned short f2bf(float f) {
    unsigned int u = __float_as_uint(f);
    u += 0x7fffu + ((u >> 16) & 1u);          // round-to-nearest-even
    return (unsigned short)(u >> 16);
}
__device__ __forceinline__ unsigned int pack2bf(float lo, float hi) {
    return (unsigned int)f2bf(lo) | ((unsigned int)f2bf(hi) << 16);
}
__device__ __forceinline__ v8f wmma_bf16(v16bf a, v16bf b, v8f c) {
    return __builtin_amdgcn_wmma_f32_16x16x32_bf16(
        /*neg_a=*/false, a, /*neg_b=*/false, b,
        /*c_mod=*/(short)0, c, /*reuse_a=*/false, /*reuse_b=*/false);
}

// ---------------------------------------------------------------------------
// Pack an [N x N] f32 matrix W into bf16 B-operand fragments for
// D = A(16x32) x B(32x16): fragment (tn, c) holds B[k][n] = W[tn*16+n][c*32+k].
// Layout: dword index = ((tn*64 + c)*32 + lane)*8 + v ; per-lane 32B chunk.
// B VGPR map (32x16, 16-bit): vgpr v<4 -> K = 8*(lane>>4)+2v, 2v+1;
// v>=4 -> K = 16 + 8*(lane>>4) + 2(v-4), +1 ; n = lane&15.
// ---------------------------------------------------------------------------
__global__ void pack_b2048_kernel(const float* __restrict__ w,
                                  unsigned int* __restrict__ dst) {
    int idx = blockIdx.x * blockDim.x + threadIdx.x;       // < 128*64*32*8
    int v    = idx & 7;
    int lane = (idx >> 3) & 31;
    int c    = (idx >> 8) & 63;
    int tn   = idx >> 14;
    int n  = tn * 16 + (lane & 15);
    int k0 = c * 32 + 8 * (lane >> 4) + (v < 4 ? 2 * v : 16 + 2 * (v - 4));
    const float* row = w + (size_t)n * RES_N + k0;
    dst[idx] = pack2bf(row[0], row[1]);
}

// W_eff = in_cor @ W_in  ([N,N]x[N,F]) packed straight into bf16 B-fragments.
__global__ void pack_win_kernel(const float* __restrict__ in_cor,
                                const float* __restrict__ w_in,
                                unsigned int* __restrict__ dst) {
    int idx = blockIdx.x * blockDim.x + threadIdx.x;       // < 128*4*32*8
    int v    = idx & 7;
    int lane = (idx >> 3) & 31;
    int c    = (idx >> 8) & 3;
    int tn   = idx >> 10;
    int n  = tn * 16 + (lane & 15);
    int f0 = c * 32 + 8 * (lane >> 4) + (v < 4 ? 2 * v : 16 + 2 * (v - 4));
    const float* ic = in_cor + (size_t)n * RES_N;
    float e0 = 0.f, e1 = 0.f;
    for (int j = 0; j < RES_N; ++j) {
        float w = ic[j];
        e0 += w * w_in[(size_t)j * FEAT_F + f0];
        e1 += w * w_in[(size_t)j * FEAT_F + f0 + 1];
    }
    dst[idx] = pack2bf(e0, e1);
}

__global__ void cvt_x_kernel(const float* __restrict__ x,
                             unsigned int* __restrict__ xb) {
    int i = blockIdx.x * blockDim.x + threadIdx.x;         // < B*T*F/2
    xb[i] = pack2bf(x[2 * i], x[2 * i + 1]);
}

__global__ void init_r_kernel(const float* __restrict__ rs,
                              float* __restrict__ rf,
                              unsigned short* __restrict__ rb) {
    int idx = blockIdx.x * blockDim.x + threadIdx.x;       // < B*N
    int n = idx & (RES_N - 1);
    float r0 = rs[n];
    rf[idx] = r0;
    rb[idx] = f2bf(r0);
}

// ---------------------------------------------------------------------------
// One recurrence step with 4-way split-K. 256 blocks x 128 threads; block b
// owns the 16x16 tile (tile_m = b>>7 of batch, tile_n = b&127 of neurons).
// Wave w (0..3) accumulates recurrent K-chunks [16w, 16w+16) plus input chunk
// w, dumps its v8f partial into LDS, then after the barrier each wave reduces
// and finishes 2 of the 8 C-rows (bias + tanh + leaky blend + stores).
// A VGPR map (16x32, 16-bit): m = lane&15; vgpr v<4 -> K = 8*(lane>>4)+2v,
// v>=4 -> K = 16 + 8*(lane>>4) + 2(v-4)  ==> two uint4 loads per chunk.
// ---------------------------------------------------------------------------
__global__ __launch_bounds__(128) void step_kernel(
    const unsigned int*  __restrict__ rb_src,   // r bf16  [B][N/2] dwords
    const float*         __restrict__ rf_src,   // r f32   [B][N]
    float*               __restrict__ rf_dst,
    unsigned short*      __restrict__ rb_dst,
    const unsigned int*  __restrict__ wrec_pack,
    const unsigned int*  __restrict__ win_pack,
    const unsigned int*  __restrict__ xb,       // x bf16 [B][T][F/2] dwords
    const float*         __restrict__ bias,
    float*               __restrict__ states,   // d_out [B][T][N]
    int t) {
    __shared__ float red[4 * 8 * 32];           // 4 partial C tiles (4 KB)

    const int lane   = threadIdx.x & 31;
    const int wv     = threadIdx.x >> 5;        // split-K part 0..3
    const int tile_m = blockIdx.x >> 7;         // 0..1
    const int tile_n = blockIdx.x & 127;        // 0..127
    const int g  = lane >> 4;
    const int ml = lane & 15;
    const int m_a = tile_m * 16 + ml;           // A-row (batch) for this lane

    v8f acc = {};

    // ---- recurrent GEMM: this wave's 16 of the 64 K-chunks -----------------
    const uint4* arow4 = (const uint4*)(rb_src + (size_t)m_a * (RES_N / 2));
    const unsigned int* bp =
        wrec_pack + (size_t)tile_n * 64 * 256 + (size_t)lane * 8;
    const int c0 = wv * 16;
    for (int i = 0; i < 16; ++i) {
        int c = c0 + i;
        Frag a;
        a.q[0] = arow4[c * 4 + g];
        a.q[1] = arow4[c * 4 + 2 + g];
        v16bf b = *(const v16bf*)(bp + c * 256);
        acc = wmma_bf16(a.bf, b, acc);
    }

    // ---- input GEMM: x_t @ W_eff^T, this wave's 1 of 4 chunks --------------
    {
        const uint4* xrow4 =
            (const uint4*)(xb + ((size_t)m_a * TIME_T + t) * (FEAT_F / 2));
        const unsigned int* bpi =
            win_pack + (size_t)tile_n * 4 * 256 + (size_t)lane * 8;
        Frag a;
        a.q[0] = xrow4[wv * 4 + g];
        a.q[1] = xrow4[wv * 4 + 2 + g];
        v16bf b = *(const v16bf*)(bpi + wv * 256);
        acc = wmma_bf16(a.bf, b, acc);
    }

    // ---- split-K reduction through LDS (lane-stride-1: conflict-free) ------
#pragma unroll
    for (int v = 0; v < 8; ++v)
        red[(wv * 8 + v) * 32 + lane] = acc[v];
    __syncthreads();

    // ---- epilogue: each wave finishes C-rows {2w, 2w+1} --------------------
    const int n = tile_n * 16 + ml;             // C: lane -> N
#pragma unroll
    for (int j = 0; j < 2; ++j) {
        int vv = wv * 2 + j;                    // C: vgpr -> M = vv + 8*g
        float s = red[(0 * 8 + vv) * 32 + lane] + red[(1 * 8 + vv) * 32 + lane]
                + red[(2 * 8 + vv) * 32 + lane] + red[(3 * 8 + vv) * 32 + lane];
        int m = tile_m * 16 + vv + 8 * g;
        float pre  = s + bias[n];
        float rold = rf_src[(size_t)m * RES_N + n];
        float rnew = (1.0f - GAMMA_F) * rold + GAMMA_F * tanhf(pre);
        rf_dst[(size_t)m * RES_N + n] = rnew;
        rb_dst[(size_t)m * RES_N + n] = f2bf(rnew);
        states[((size_t)m * TIME_T + t) * RES_N + n] = rnew;
    }
}

// ---------------------------------------------------------------------------
// Finale: OUT = STATES @ out_cor^T over all B*T = 16384 rows, in place.
// Each block stages its 16-row stripe into LDS as bf16 (64 KB of the WGP's
// 320 KB), syncs, then 8 waves run LDS-A x packed-B WMMAs and overwrite.
// ---------------------------------------------------------------------------
__global__ __launch_bounds__(256) void outcor_kernel(
    float* __restrict__ out, const unsigned int* __restrict__ ocp) {
    __shared__ unsigned int lds[16 * (RES_N / 2)];     // 16 rows bf16 = 64 KB
    float* rows = out + (size_t)blockIdx.x * 16 * RES_N;

    for (int i = threadIdx.x; i < 16 * (RES_N / 2); i += 256)
        lds[i] = pack2bf(rows[2 * i], rows[2 * i + 1]);
    __syncthreads();

    const int lane = threadIdx.x & 31;
    const int wv   = threadIdx.x >> 5;                 // 8 waves
    const int g  = lane >> 4;
    const int ml = lane & 15;
    const uint4* arow4 = (const uint4*)(lds + (size_t)ml * (RES_N / 2));

    for (int jb = 0; jb < 4; ++jb) {                   // 4 groups of 4 tiles
        const int tn0 = wv * 16 + jb * 4;
        const unsigned int* bp0 = ocp + (size_t)(tn0 + 0) * 16384 + lane * 8;
        const unsigned int* bp1 = ocp + (size_t)(tn0 + 1) * 16384 + lane * 8;
        const unsigned int* bp2 = ocp + (size_t)(tn0 + 2) * 16384 + lane * 8;
        const unsigned int* bp3 = ocp + (size_t)(tn0 + 3) * 16384 + lane * 8;
        v8f a0 = {}, a1 = {}, a2 = {}, a3 = {};
        for (int c = 0; c < 64; ++c) {
            Frag a;
            a.q[0] = arow4[c * 4 + g];                 // ds_load_b128
            a.q[1] = arow4[c * 4 + 2 + g];
            a0 = wmma_bf16(a.bf, *(const v16bf*)(bp0 + c * 256), a0);
            a1 = wmma_bf16(a.bf, *(const v16bf*)(bp1 + c * 256), a1);
            a2 = wmma_bf16(a.bf, *(const v16bf*)(bp2 + c * 256), a2);
            a3 = wmma_bf16(a.bf, *(const v16bf*)(bp3 + c * 256), a3);
        }
#pragma unroll
        for (int v = 0; v < 8; ++v) {
            int m = v + 8 * g;
            rows[(size_t)m * RES_N + (tn0 + 0) * 16 + ml] = a0[v];
            rows[(size_t)m * RES_N + (tn0 + 1) * 16 + ml] = a1[v];
            rows[(size_t)m * RES_N + (tn0 + 2) * 16 + ml] = a2[v];
            rows[(size_t)m * RES_N + (tn0 + 3) * 16 + ml] = a3[v];
        }
    }
}

// ---------------------------------------------------------------------------
extern "C" void kernel_launch(void* const* d_in, const int* in_sizes, int n_in,
                              void* d_out, int out_size, void* d_ws,
                              size_t ws_size, hipStream_t stream) {
    const float* x      = (const float*)d_in[0];   // [B,T,F]
    const float* w_in   = (const float*)d_in[1];   // [N,F]
    const float* w_rec  = (const float*)d_in[2];   // [N,N]
    const float* bias   = (const float*)d_in[3];   // [N]
    const float* rstart = (const float*)d_in[4];   // [N]
    const float* in_cor = (const float*)d_in[5];   // [N,N]
    const float* o_cor  = (const float*)d_in[6];   // [N,N]
    float* out = (float*)d_out;

    unsigned char* ws = (unsigned char*)d_ws;
    size_t off = 0;
    unsigned int* wrec_pack = (unsigned int*)(ws + off); off += (size_t)8 << 20;
    unsigned int* ocor_pack = (unsigned int*)(ws + off); off += (size_t)8 << 20;
    unsigned int* win_pack  = (unsigned int*)(ws + off); off += (size_t)512 << 10;
    unsigned int* xb        = (unsigned int*)(ws + off); off += (size_t)4 << 20;
    float* rf[2];
    rf[0] = (float*)(ws + off); off += (size_t)BATCH_B * RES_N * 4;
    rf[1] = (float*)(ws + off); off += (size_t)BATCH_B * RES_N * 4;
    unsigned short* rb[2];
    rb[0] = (unsigned short*)(ws + off); off += (size_t)BATCH_B * RES_N * 2;
    rb[1] = (unsigned short*)(ws + off); off += (size_t)BATCH_B * RES_N * 2;
    (void)ws_size; (void)in_sizes; (void)n_in; (void)out_size;

    // ---- prep (regenerated every call: deterministic) ----------------------
    pack_b2048_kernel<<<(128 * 64 * 32 * 8) / 256, 256, 0, stream>>>(w_rec,
                                                                     wrec_pack);
    pack_b2048_kernel<<<(128 * 64 * 32 * 8) / 256, 256, 0, stream>>>(o_cor,
                                                                     ocor_pack);
    pack_win_kernel<<<(128 * 4 * 32 * 8) / 256, 256, 0, stream>>>(in_cor, w_in,
                                                                  win_pack);
    cvt_x_kernel<<<(BATCH_B * TIME_T * FEAT_F / 2) / 256, 256, 0, stream>>>(x,
                                                                            xb);
    init_r_kernel<<<(BATCH_B * RES_N) / 256, 256, 0, stream>>>(rstart, rf[0],
                                                               rb[0]);

    // ---- sequential recurrence: one launch per step is the device-wide sync
    for (int t = 0; t < TIME_T; ++t) {
        int s = t & 1, d = s ^ 1;
        step_kernel<<<256, 128, 0, stream>>>(
            (const unsigned int*)rb[s], rf[s], rf[d], rb[d],
            wrec_pack, win_pack, xb, bias, out, t);
    }

    // ---- out = states @ out_cor^T (parallel, in place) ---------------------
    outcor_kernel<<<(BATCH_B * TIME_T) / 16, 256, 0, stream>>>(out, ocor_pack);
}
